// Block_22591527977234
// MI455X (gfx1250) — matrix operand order repaired and checked by reading, not compiled, not run on hardware
//
#include <hip/hip_runtime.h>

// ---------------------------------------------------------------------------
// Transformer block forward for MI455X (gfx1250), wave32 + WMMA f16 + TDM.
// B=4096, T=32, C=180, H=6, HD=30.  All GEMMs via v_wmma_f32_16x16x32_f16.
// x tiles staged into LDS with TENSOR_LOAD_TO_LDS (Tensor Data Mover).
// ---------------------------------------------------------------------------

typedef __attribute__((ext_vector_type(16))) _Float16 v16h;
typedef __attribute__((ext_vector_type(8)))  _Float16 v8h;
typedef __attribute__((ext_vector_type(8)))  float    v8f;
typedef __attribute__((ext_vector_type(4)))  unsigned tdm_u4;
typedef __attribute__((ext_vector_type(8)))  int      tdm_i8;
typedef __attribute__((ext_vector_type(4)))  int      tdm_i4;

static constexpr int kB  = 4096;
static constexpr int kT  = 32;
static constexpr int kC  = 180;
static constexpr int kH  = 6;
static constexpr int kHD = 30;
static constexpr int kF  = 4 * kC;   // 720

// packed fragment sizes (halves)
static constexpr int kQKVPackHalves = 3 * kH * 6 * 2 * 512;  // 110592
static constexpr int kW1PackHalves  = 6 * 45 * 512;          // 138240
static constexpr int kW2PackHalves  = 23 * 12 * 512;         // 141312

// ---------------------------------------------------------------------------
// Tensor Data Mover: 1-D DMA of `ndwords` dwords global -> LDS.
// D# layout per CDNA5 ISA (group0: flags/lds/global addr,
// group1: data_size=4B, tensor_dim0 = tile_dim0 = ndwords, 1 row).
// Issue from a single wave; EXEC is ignored by TDM.
// ---------------------------------------------------------------------------
__device__ inline void tdm_load_1d(const void* gptr, unsigned lds_off,
                                   unsigned ndwords) {
  unsigned long long ga = (unsigned long long)gptr;
  tdm_u4 g0;
  g0[0] = 1u;                                   // count=1 (valid), user mode
  g0[1] = lds_off;                              // lds_addr (bytes)
  g0[2] = (unsigned)ga;                         // global_addr[31:0]
  g0[3] = (unsigned)(ga >> 32) | 0x80000000u;   // global_addr[56:32] | type=2
  tdm_i8 g1;
  g1[0] = 0x00020000;                           // data_size=2 -> 4 bytes
  g1[1] = (int)(ndwords << 16);                 // tensor_dim0[15:0] @ [31:16]
  g1[2] = (int)((ndwords >> 16) & 0xffffu) | (1 << 16);  // dim0 hi | dim1=1
  g1[3] = (int)(ndwords << 16);                 // tile_dim0 @ [31:16]
  g1[4] = 1;                                    // tile_dim1 = 1
  g1[5] = (int)ndwords;                         // tensor_dim0_stride lo
  g1[6] = 0;
  g1[7] = 0;
  tdm_i4 z = {};
#if __clang_major__ >= 23
  tdm_i8 z8 = {};
  __builtin_amdgcn_tensor_load_to_lds(g0, g1, z, z, z8, 0);
#else
  __builtin_amdgcn_tensor_load_to_lds(g0, g1, z, z, 0);
#endif
}

__device__ inline unsigned lds_offset_of(const void* p) {
  // Generic pointers to LDS carry the byte offset in the low 32 bits.
  return (unsigned)(unsigned long long)p;
}

// ---------------------------------------------------------------------------
// Fragment helpers (A/B in contiguous k order, k = 16*(lane>>4) + e; C/D per CDNA5 ISA 7.12.2)
// A 16x32 f16: lanes 0-15 = rows M0-15 holding K 0..15;
//              lanes 16-31 = same rows holding K 16..31.
// B 32x16 f16: lane&15 = N column; lanes 0-15 hold K0-15, lanes16-31 K16-31.
// C/D f32: lane&15 = N; vgpr r = row M = r (+8 for lanes 16-31).
// ---------------------------------------------------------------------------
__device__ inline v16h cat16(v8h lo, v8h hi) {
  return __builtin_shufflevector(lo, hi, 0,1,2,3,4,5,6,7,8,9,10,11,12,13,14,15);
}

// A fragment from row-major f16 matrix in LDS. base -> [row0][k0] of tile.
__device__ inline v16h load_a_frag(const _Float16* base, int stride, int lane) {
  const int row = lane & 15;
  const int hi  = lane >> 4;
  const _Float16* p = base + row * stride + hi * 16;
  return cat16(*(const v8h*)p, *(const v8h*)(p + 8));
}

// B fragment from [n][k] (K-contiguous per column) f16 storage in LDS.
__device__ inline v16h load_b_frag_nk(const _Float16* base, int stride, int lane) {
  const int n  = lane & 15;
  const int hi = lane >> 4;
  const _Float16* p = base + n * stride + hi * 16;
  return cat16(*(const v8h*)p, *(const v8h*)(p + 8));
}

__device__ inline v8f wmma_f16(v16h a, v16h b, v8f c) {
  return __builtin_amdgcn_wmma_f32_16x16x32_f16(
      /*neg_a=*/false, a, /*neg_b=*/false, b,
      /*c_mod=*/(short)0, c, /*reuse_a=*/false, /*reuse_b=*/false);
}

// ---------------------------------------------------------------------------
// Weight packing: fp32 KxN row-major -> f16 B-fragment layout.
// dst[(kt*ntiles + nt)*512 + lane*16 + e], k = kt*32 + (lane>=16)*16 + e,
// n = nt*16 + (lane&15).  Out-of-range -> 0 (zero padding).
// ---------------------------------------------------------------------------
__global__ void pack_b16(const float* __restrict__ src, _Float16* __restrict__ dst,
                         int K, int N, int ntiles, int count) {
  int idx = blockIdx.x * blockDim.x + threadIdx.x;
  if (idx >= count) return;
  int e    = idx & 15;
  int lane = (idx >> 4) & 31;
  int t    = idx >> 9;
  int nt   = t % ntiles;
  int kt   = t / ntiles;
  int k = kt * 32 + ((lane >> 4) << 4) + e;
  int n = nt * 16 + (lane & 15);
  float v = (k < K && n < N) ? src[(size_t)k * N + n] : 0.0f;
  dst[idx] = (_Float16)v;
}

// ---------------------------------------------------------------------------
// Kernel 1: LN1 + multi-head causal attention + residual.  1 block / batch.
// ---------------------------------------------------------------------------
__global__ __launch_bounds__(128) void attn_block_kernel(
    const float* __restrict__ x, const float* __restrict__ g1,
    const float* __restrict__ be1, const _Float16* __restrict__ wqkvp,
    float* __restrict__ out) {
  __shared__ float    xbuf[kT][kC];        // raw x rows (for residual)
  __shared__ _Float16 hbuf[kT][192];       // LN1 output, K padded to 192
  __shared__ _Float16 qbuf[kH][kT][32];    // q row-major [t][d]
  __shared__ _Float16 kbuf[kH][kT][32];    // k row-major [s][d] ([n][k] for scores)
  __shared__ _Float16 vTbf[kH][32][kT];    // v transposed [d][s] ([n][k] for out)
  __shared__ float    sbuf[kH][kT][kT];    // scores f32
  __shared__ _Float16 abuf[kH][kT][kT];    // softmax probs f16

  const int tid  = threadIdx.x;
  const int lane = tid & 31;
  const int wave = tid >> 5;
  const int b    = blockIdx.x;
  const float* xb = x + (size_t)b * kT * kC;

  // ---- stage x tile via Tensor Data Mover (one wave issues) ----
  if (tid < 32) {
    tdm_load_1d(xb, lds_offset_of(&xbuf[0][0]), kT * kC);
    __builtin_amdgcn_s_wait_tensorcnt(0);
  }
  __syncthreads();

  // ---- LayerNorm 1 (4 threads per row) ----
  {
    const int row = tid >> 2, q = tid & 3;
    float s = 0.f, ss = 0.f;
    for (int c = q; c < kC; c += 4) { float v = xbuf[row][c]; s += v; ss += v * v; }
    s  += __shfl_xor(s, 1);  s  += __shfl_xor(s, 2);
    ss += __shfl_xor(ss, 1); ss += __shfl_xor(ss, 2);
    const float mu   = s * (1.0f / kC);
    const float var  = ss * (1.0f / kC) - mu * mu;
    const float rinv = rsqrtf(var + 1e-5f);
    for (int c = q; c < 192; c += 4) {
      float hv = (c < kC) ? ((xbuf[row][c] - mu) * rinv * g1[c] + be1[c]) : 0.0f;
      hbuf[row][c] = (_Float16)hv;
    }
  }
  __syncthreads();

  // ---- QKV projections: per matrix, 6 heads * 2 Mtiles * 2 Ntiles = 24 ----
  const int n_ = lane & 15, hi_ = lane >> 4;

  // Q
  for (int tile = wave; tile < 24; tile += 4) {
    const int head = tile >> 2, mt = (tile >> 1) & 1, nt = tile & 1;
    v8f acc = {};
#pragma unroll
    for (int kt = 0; kt < 6; ++kt) {
      v16h a  = load_a_frag(&hbuf[mt * 16][kt * 32], 192, lane);
      v16h bf = *(const v16h*)(wqkvp + ((size_t)(((0 * kH + head) * 6 + kt) * 2 + nt) * 512 + lane * 16));
      acc = wmma_f16(a, bf, acc);
    }
#pragma unroll
    for (int r = 0; r < 8; ++r)
      qbuf[head][mt * 16 + r + hi_ * 8][nt * 16 + n_] = (_Float16)acc[r];
  }
  // K
  for (int tile = wave; tile < 24; tile += 4) {
    const int head = tile >> 2, mt = (tile >> 1) & 1, nt = tile & 1;
    v8f acc = {};
#pragma unroll
    for (int kt = 0; kt < 6; ++kt) {
      v16h a  = load_a_frag(&hbuf[mt * 16][kt * 32], 192, lane);
      v16h bf = *(const v16h*)(wqkvp + ((size_t)(((1 * kH + head) * 6 + kt) * 2 + nt) * 512 + lane * 16));
      acc = wmma_f16(a, bf, acc);
    }
#pragma unroll
    for (int r = 0; r < 8; ++r)
      kbuf[head][mt * 16 + r + hi_ * 8][nt * 16 + n_] = (_Float16)acc[r];
  }
  // V (stored transposed; 8 consecutive rows per lane -> one 16B store)
  for (int tile = wave; tile < 24; tile += 4) {
    const int head = tile >> 2, mt = (tile >> 1) & 1, nt = tile & 1;
    v8f acc = {};
#pragma unroll
    for (int kt = 0; kt < 6; ++kt) {
      v16h a  = load_a_frag(&hbuf[mt * 16][kt * 32], 192, lane);
      v16h bf = *(const v16h*)(wqkvp + ((size_t)(((2 * kH + head) * 6 + kt) * 2 + nt) * 512 + lane * 16));
      acc = wmma_f16(a, bf, acc);
    }
    v8h hv;
#pragma unroll
    for (int r = 0; r < 8; ++r) hv[r] = (_Float16)acc[r];
    *(v8h*)&vTbf[head][nt * 16 + n_][mt * 16 + hi_ * 8] = hv;
  }
  __syncthreads();

  // ---- scores = q @ k^T : 6 heads * 2x2 tiles, K=32 (single WMMA) ----
  for (int tile = wave; tile < 24; tile += 4) {
    const int head = tile >> 2, mt = (tile >> 1) & 1, nt = tile & 1;
    v16h a  = load_a_frag(&qbuf[head][mt * 16][0], 32, lane);
    v16h bf = load_b_frag_nk(&kbuf[head][nt * 16][0], 32, lane);
    v8f acc = {};
    acc = wmma_f16(a, bf, acc);
#pragma unroll
    for (int r = 0; r < 8; ++r)
      sbuf[head][mt * 16 + r + hi_ * 8][nt * 16 + n_] = acc[r];
  }
  __syncthreads();

  // ---- causal softmax (per row, 192 rows over 128 threads) ----
  {
    const float scale = 0.1825741858f;  // 1/sqrt(30)
    for (int rr = tid; rr < kH * kT; rr += 128) {
      const int head = rr >> 5, t = rr & 31;
      float m = -1e30f;
      for (int s = 0; s <= t; ++s) m = fmaxf(m, sbuf[head][t][s] * scale);
      float sum = 0.f;
      for (int s = 0; s <= t; ++s) sum += __expf(sbuf[head][t][s] * scale - m);
      const float inv = 1.0f / sum;
      for (int s = 0; s < kT; ++s) {
        float p = (s <= t) ? __expf(sbuf[head][t][s] * scale - m) * inv : 0.0f;
        abuf[head][t][s] = (_Float16)p;
      }
    }
  }
  __syncthreads();

  // ---- out = attn @ v, fused residual store ----
  for (int tile = wave; tile < 24; tile += 4) {
    const int head = tile >> 2, mt = (tile >> 1) & 1, nt = tile & 1;
    v16h a  = load_a_frag(&abuf[head][mt * 16][0], 32, lane);
    v16h bf = load_b_frag_nk(&vTbf[head][nt * 16][0], 32, lane);
    v8f acc = {};
    acc = wmma_f16(a, bf, acc);
    const int d = nt * 16 + n_;
    if (d < kHD) {
      const int col = head * kHD + d;
#pragma unroll
      for (int r = 0; r < 8; ++r) {
        const int trow = mt * 16 + r + hi_ * 8;
        out[((size_t)b * kT + trow) * kC + col] = xbuf[trow][col] + acc[r];
      }
    }
  }
}

// ---------------------------------------------------------------------------
// Kernel 2: LN2 + MLP (180 -> 720 -> 180, ReLU) + residual, in-place on d_out.
// 1 block = 32 token rows, 8 waves. 4C intermediate lives entirely in LDS.
// ---------------------------------------------------------------------------
__global__ __launch_bounds__(256) void mlp_block_kernel(
    float* __restrict__ xio, const float* __restrict__ g2,
    const float* __restrict__ be2, const _Float16* __restrict__ w1p,
    const float* __restrict__ b1, const _Float16* __restrict__ w2p,
    const float* __restrict__ b2) {
  __shared__ float    xb[kT][kC];      // x1 rows (residual)
  __shared__ _Float16 hb[kT][192];     // LN2 output, K padded
  __shared__ _Float16 yb[kT][736];     // ReLU(h@w1+b1), K padded to 736

  const int tid  = threadIdx.x;
  const int lane = tid & 31;
  const int wave = tid >> 5;
  const size_t rowbase = (size_t)blockIdx.x * kT;

  // ---- stage x1 tile via TDM, zero-pad yb tail columns ----
  if (tid < 32) {
    tdm_load_1d(xio + rowbase * kC, lds_offset_of(&xb[0][0]), kT * kC);
    __builtin_amdgcn_s_wait_tensorcnt(0);
  }
  for (int i = tid; i < kT * 16; i += 256) yb[i >> 4][720 + (i & 15)] = (_Float16)0.0f;
  __syncthreads();

  // ---- LayerNorm 2 (8 threads per row) ----
  {
    const int row = tid >> 3, q = tid & 7;
    float s = 0.f, ss = 0.f;
    for (int c = q; c < kC; c += 8) { float v = xb[row][c]; s += v; ss += v * v; }
    s  += __shfl_xor(s, 1);  s  += __shfl_xor(s, 2);  s  += __shfl_xor(s, 4);
    ss += __shfl_xor(ss, 1); ss += __shfl_xor(ss, 2); ss += __shfl_xor(ss, 4);
    const float mu   = s * (1.0f / kC);
    const float var  = ss * (1.0f / kC) - mu * mu;
    const float rinv = rsqrtf(var + 1e-5f);
    for (int c = q; c < 192; c += 8) {
      float hv = (c < kC) ? ((xb[row][c] - mu) * rinv * g2[c] + be2[c]) : 0.0f;
      hb[row][c] = (_Float16)hv;
    }
  }
  __syncthreads();

  // ---- GEMM1: (32x192) @ (192x720) + b1, ReLU -> yb.  90 tiles / 8 waves ----
  for (int tile = wave; tile < 90; tile += 8) {
    const int mt = tile / 45;
    const int nt = tile % 45;
    v8f acc = {};
#pragma unroll
    for (int kt = 0; kt < 6; ++kt) {
      v16h a  = load_a_frag(&hb[mt * 16][kt * 32], 192, lane);
      v16h bf = *(const v16h*)(w1p + ((size_t)(kt * 45 + nt) * 512 + lane * 16));
      acc = wmma_f16(a, bf, acc);
    }
    const int n = lane & 15, hi = lane >> 4;
    const float bias = b1[nt * 16 + n];
#pragma unroll
    for (int r = 0; r < 8; ++r)
      yb[mt * 16 + r + hi * 8][nt * 16 + n] = (_Float16)fmaxf(acc[r] + bias, 0.0f);
  }
  __syncthreads();

  // ---- GEMM2: (32x736) @ (736x192) + b2 + residual -> global.  24 tiles ----
  for (int tile = wave; tile < 24; tile += 8) {
    const int mt = tile / 12;
    const int nt = tile % 12;
    v8f acc = {};
    for (int kt = 0; kt < 23; ++kt) {
      v16h a  = load_a_frag(&yb[mt * 16][kt * 32], 736, lane);
      v16h bf = *(const v16h*)(w2p + ((size_t)(kt * 12 + nt) * 512 + lane * 16));
      acc = wmma_f16(a, bf, acc);
    }
    const int n = lane & 15, hi = lane >> 4;
    const int col = nt * 16 + n;
    if (col < kC) {
      const float bias = b2[col];
#pragma unroll
      for (int r = 0; r < 8; ++r) {
        const int row = mt * 16 + r + hi * 8;
        xio[(rowbase + row) * kC + col] = xb[row][col] + acc[r] + bias;
      }
    }
  }
}

// ---------------------------------------------------------------------------
// Host launcher
// ---------------------------------------------------------------------------
extern "C" void kernel_launch(void* const* d_in, const int* in_sizes, int n_in,
                              void* d_out, int out_size, void* d_ws, size_t ws_size,
                              hipStream_t stream) {
  const float* x   = (const float*)d_in[0];
  const float* wq  = (const float*)d_in[1];
  const float* wk  = (const float*)d_in[2];
  const float* wv  = (const float*)d_in[3];
  const float* g1  = (const float*)d_in[4];
  const float* be1 = (const float*)d_in[5];
  const float* g2  = (const float*)d_in[6];
  const float* be2 = (const float*)d_in[7];
  const float* w1  = (const float*)d_in[8];
  const float* b1  = (const float*)d_in[9];
  const float* w2  = (const float*)d_in[10];
  const float* b2  = (const float*)d_in[11];
  float* out = (float*)d_out;

  _Float16* wqkvp = (_Float16*)d_ws;
  _Float16* w1p   = wqkvp + kQKVPackHalves;
  _Float16* w2p   = w1p + kW1PackHalves;

  // Pack weights to f16 B-fragment layout (per head for qkv).
  const float* wmats[3] = {wq, wk, wv};
  const int qkv_cnt = 6 * 2 * 512;  // per (mat, head)
  for (int mat = 0; mat < 3; ++mat)
    for (int head = 0; head < kH; ++head)
      pack_b16<<<(qkv_cnt + 255) / 256, 256, 0, stream>>>(
          wmats[mat] + (size_t)head * kC * kHD,
          wqkvp + (size_t)(mat * kH + head) * qkv_cnt, kC, kHD, 2, qkv_cnt);

  pack_b16<<<(kW1PackHalves + 255) / 256, 256, 0, stream>>>(w1, w1p, kC, kF, 45,
                                                            kW1PackHalves);
  pack_b16<<<(kW2PackHalves + 255) / 256, 256, 0, stream>>>(w2, w2p, kF, kC, 12,
                                                            kW2PackHalves);

  // Attention (writes x1 = x + attn to d_out), then MLP in-place on d_out.
  attn_block_kernel<<<kB, 128, 0, stream>>>(x, g1, be1, wqkvp, out);
  mlp_block_kernel<<<kB, 256, 0, stream>>>(out, g2, be2, w1p, b1, w2p, b2);
}
